// EgoGNN_19473381720719
// MI455X (gfx1250) — matrix operand order repaired and checked by reading, not compiled, not run on hardware
//
#include <hip/hip_runtime.h>
#include <hip/hip_bf16.h>

// ---------------------------------------------------------------------------
// CDNA5 (gfx1250) EgoGNN pipeline, built around v_wmma_f32_16x16x32_bf16.
// All matmuls (per-node 16x16 ego products and the dense linears) go through
// WMMA with fp32 accumulators; graph propagation is L2-resident f32 atomics.
// ---------------------------------------------------------------------------

typedef __attribute__((ext_vector_type(16))) __bf16 bf16x16;
typedef __attribute__((ext_vector_type(8)))  __bf16 bf16x8;
typedef __attribute__((ext_vector_type(8)))  float  f32x8;

__device__ __forceinline__ __bf16 f2bf(float f) {
    return (__bf16)f;    // native f32->bf16 convert (RNE) on gfx1250
}

// ---------------------------------------------------------------------------
// degree helpers
// ---------------------------------------------------------------------------
__global__ void deg_init_kernel(float* deg, int n) {
    int i = blockIdx.x * blockDim.x + threadIdx.x;
    if (i < n) deg[i] = 1.0f;                       // self loop
}

__global__ void deg_count_kernel(const int* __restrict__ dst, float* deg, int e) {
    int i = blockIdx.x * blockDim.x + threadIdx.x;
    if (i < e) atomicAdd(&deg[dst[i]], 1.0f);
}

__global__ void deg_rsqrt_kernel(const float* __restrict__ deg, float* dinv, int n) {
    int i = blockIdx.x * blockDim.x + threadIdx.x;
    if (i < n) dinv[i] = deg[i] > 0.0f ? rsqrtf(deg[i]) : 0.0f;
}

__global__ void zero_kernel(float* p, size_t count) {
    size_t i = (size_t)blockIdx.x * blockDim.x + threadIdx.x;
    if (i < count) p[i] = 0.0f;
}

// ---------------------------------------------------------------------------
// ego_conv: one wave per node.  y = A*(A*xg); scatter-add y into out.
// A: [16x16] fp32 (zero-padded to 16x32 bf16 A-operand).
// xg staged transposed in LDS as bf16 [128 feat][16 k].
// Lanes 16-31 (K=16..31 padding half) read their B operand from a zeroed
// LDS pad with stride 0 -> all ds loads are branch-free.
// ---------------------------------------------------------------------------
__global__ __launch_bounds__(128) void ego_conv_kernel(
    const float* __restrict__ x, const int* __restrict__ ego_ids,
    const float* __restrict__ adj, float* __restrict__ out, int n_nodes)
{
    __shared__ __align__(16) __bf16 lds_t[4][128 * 16];  // transposed tiles
    __shared__ int    lds_ids[4][16];
    __shared__ __align__(16) __bf16 zpad[16];            // zero K-pad source

    const int wave = threadIdx.x >> 5;
    const int lane = threadIdx.x & 31;
    const int node = blockIdx.x * 4 + wave;
    const bool active = node < n_nodes;

    __bf16* xt  = lds_t[wave];
    int*    ids = lds_ids[wave];

    if (threadIdx.x < 16) zpad[threadIdx.x] = __builtin_bit_cast(__bf16, (unsigned short)0);
    if (active && lane < 16) ids[lane] = ego_ids[node * 16 + lane];
    __syncthreads();

    // prefetch this node's adjacency (1KB) while we gather features
    if (active) __builtin_prefetch(adj + (size_t)node * 256, 0, 0);

    // gather x[ids[k]] -> LDS transposed bf16 (each lane owns 4 features)
    if (active) {
        const int j0 = lane * 4;
        for (int k = 0; k < 16; ++k) {
            const float4 v = *reinterpret_cast<const float4*>(
                x + (size_t)ids[k] * 128 + j0);
            xt[(j0 + 0) * 16 + k] = f2bf(v.x);
            xt[(j0 + 1) * 16 + k] = f2bf(v.y);
            xt[(j0 + 2) * 16 + k] = f2bf(v.z);
            xt[(j0 + 3) * 16 + k] = f2bf(v.w);
        }
    }
    __syncthreads();

    // A operand (16x16 zero-padded to 16x32):
    //   lanes 0-15: elems 0-15 = K0-15
    //   lanes16-31: elems 0-15 = K16-31 (zero)
    bf16x16 a = {};
    const int arow = lane & 15;
    if (active && lane < 16) {
        const float* ap = adj + (size_t)node * 256 + arow * 16;
        const float4 a0 = *reinterpret_cast<const float4*>(ap);
        const float4 a1 = *reinterpret_cast<const float4*>(ap + 4);
        const float4 a2 = *reinterpret_cast<const float4*>(ap + 8);
        const float4 a3 = *reinterpret_cast<const float4*>(ap + 12);
        a[0] = f2bf(a0.x); a[1] = f2bf(a0.y); a[2] = f2bf(a0.z); a[3] = f2bf(a0.w);
        a[4] = f2bf(a1.x); a[5] = f2bf(a1.y); a[6] = f2bf(a1.z); a[7] = f2bf(a1.w);
        a[8] = f2bf(a2.x); a[9] = f2bf(a2.y); a[10] = f2bf(a2.z); a[11] = f2bf(a2.w);
        a[12] = f2bf(a3.x); a[13] = f2bf(a3.y); a[14] = f2bf(a3.z); a[15] = f2bf(a3.w);
    }

    const int col = lane & 15;
    const int mhi = (lane < 16) ? 0 : 8;
    // branch-free B addressing: padding lanes read zeros with stride 0
    const __bf16* bbase = (lane < 16) ? (xt + col * 16) : zpad;
    const int     bstep = (lane < 16) ? 256 : 0;

    f32x8 zero8 = {};
    f32x8 acc[8];
    bf16x16 bt[8];

    // ---- power 1: acc = A * xg -------------------------------------------
#pragma unroll
    for (int t = 0; t < 8; ++t) acc[t] = zero8;
#pragma unroll
    for (int t = 0; t < 8; ++t) {            // batch all ds loads
        const __bf16* bp = bbase + t * bstep;
#pragma unroll
        for (int e = 0; e < 16; ++e) bt[t][e] = bp[e];
    }
#pragma unroll
    for (int t = 0; t < 8; ++t)              // back-to-back WMMAs
        acc[t] = __builtin_amdgcn_wmma_f32_16x16x32_bf16(
            false, a, false, bt[t], (short)0, acc[t], false, false);

    // ---- write y1 back to LDS (transposed, bf16, packed 16B stores) ------
    __syncthreads();
    if (active) {
#pragma unroll
        for (int t = 0; t < 8; ++t) {
            bf16x8 pk;
#pragma unroll
            for (int r = 0; r < 8; ++r) pk[r] = f2bf(acc[t][r]);
            *reinterpret_cast<bf16x8*>(xt + (t * 16 + col) * 16 + mhi) = pk;
        }
    }
    __syncthreads();

    // ---- power 2: acc = A * y1 -------------------------------------------
#pragma unroll
    for (int t = 0; t < 8; ++t) acc[t] = zero8;
#pragma unroll
    for (int t = 0; t < 8; ++t) {
        const __bf16* bp = bbase + t * bstep;
#pragma unroll
        for (int e = 0; e < 16; ++e) bt[t][e] = bp[e];
    }
#pragma unroll
    for (int t = 0; t < 8; ++t)
        acc[t] = __builtin_amdgcn_wmma_f32_16x16x32_bf16(
            false, a, false, bt[t], (short)0, acc[t], false, false);

    // ---- scatter-add into out[ids[M]] ------------------------------------
    if (active) {
#pragma unroll
        for (int r = 0; r < 8; ++r) {
            size_t orow = (size_t)ids[mhi + r] * 128;
#pragma unroll
            for (int t = 0; t < 8; ++t)
                atomicAdd(out + orow + t * 16 + col, acc[t][r]);
        }
    }
}

// ---------------------------------------------------------------------------
// Dense GEMM: out[N,FOUT] = act(rowscale(A[N,128]) @ W[128,FOUT] (+ bias))
// W staged transposed in LDS as bf16 so each lane's B operand is two contiguous 8-element k runs.
// One wave computes a 16-row by FOUT-col stripe.  Per k-chunk, all B tiles
// are preloaded before the WMMA chain.
// ---------------------------------------------------------------------------
template <int FOUT>
__global__ __launch_bounds__(128) void gemm_kernel(
    const float* __restrict__ A, const float* __restrict__ W,
    const float* __restrict__ bias, const float* __restrict__ rowscale,
    float* __restrict__ out, int n_rows, int relu)
{
    __shared__ __align__(16) __bf16 wt[FOUT * 128];   // wt[c*128 + r] = W[r][c]
    for (int idx = threadIdx.x; idx < 128 * FOUT; idx += 128) {
        int r = idx / FOUT, c = idx % FOUT;     // FOUT is a power of two
        wt[c * 128 + r] = f2bf(W[idx]);
    }
    __syncthreads();

    const int wave = threadIdx.x >> 5;
    const int lane = threadIdx.x & 31;
    const int rb   = blockIdx.x * 4 + wave;     // 16-row block index
    if (rb * 16 >= n_rows) return;              // no barriers past this point

    constexpr int NT = FOUT / 16;
    const int col = lane & 15;
    const int klo = (lane < 16) ? 0 : 8;
    const int khi = (lane < 16) ? 0 : 16;
    const int row = rb * 16 + col;              // A-operand row (M = lane&15)
    const float* arow  = A + (size_t)row * 128;
    const float  scale = rowscale ? rowscale[row] : 1.0f;

    f32x8 acc[NT] = {};
    bf16x16 bt[NT];

    for (int kc = 0; kc < 128; kc += 32) {
        const float4 a0 = *reinterpret_cast<const float4*>(arow + kc + klo);
        const float4 a1 = *reinterpret_cast<const float4*>(arow + kc + klo + 4);
        const float4 a2 = *reinterpret_cast<const float4*>(arow + kc + 16 + klo);
        const float4 a3 = *reinterpret_cast<const float4*>(arow + kc + 16 + klo + 4);
        bf16x16 a;
        a[0]  = f2bf(a0.x * scale); a[1]  = f2bf(a0.y * scale);
        a[2]  = f2bf(a0.z * scale); a[3]  = f2bf(a0.w * scale);
        a[4]  = f2bf(a1.x * scale); a[5]  = f2bf(a1.y * scale);
        a[6]  = f2bf(a1.z * scale); a[7]  = f2bf(a1.w * scale);
        a[8]  = f2bf(a2.x * scale); a[9]  = f2bf(a2.y * scale);
        a[10] = f2bf(a2.z * scale); a[11] = f2bf(a2.w * scale);
        a[12] = f2bf(a3.x * scale); a[13] = f2bf(a3.y * scale);
        a[14] = f2bf(a3.z * scale); a[15] = f2bf(a3.w * scale);
#pragma unroll
        for (int t = 0; t < NT; ++t) {          // batch all ds loads
            const __bf16* bp = wt + (t * 16 + col) * 128 + kc + klo;
#pragma unroll
            for (int e = 0; e < 16; ++e) bt[t][e] = bp[(e < 8) ? e : e + 8];
        }
#pragma unroll
        for (int t = 0; t < NT; ++t)            // back-to-back WMMAs
            acc[t] = __builtin_amdgcn_wmma_f32_16x16x32_bf16(
                false, a, false, bt[t], (short)0, acc[t], false, false);
    }

    const int mhi = (lane < 16) ? 0 : 8;
#pragma unroll
    for (int t = 0; t < NT; ++t) {
        float bv = bias ? bias[t * 16 + col] : 0.0f;
#pragma unroll
        for (int r = 0; r < 8; ++r) {
            float v = acc[t][r] + bv;
            if (relu) v = fmaxf(v, 0.0f);
            out[(size_t)(rb * 16 + mhi + r) * FOUT + t * 16 + col] = v;
        }
    }
}

// ---------------------------------------------------------------------------
// GCN propagation: out[n] = bias + dinv[n]^2 * h[n]  (init)
//                  out[d] += dinv[s]*dinv[d]*h[s]    (edge scatter)
// ---------------------------------------------------------------------------
__global__ void gcn_init_kernel(const float* __restrict__ h,
                                const float* __restrict__ bias,
                                const float* __restrict__ dinv,
                                float* __restrict__ out, size_t total, int fshift) {
    size_t i = (size_t)blockIdx.x * blockDim.x + threadIdx.x;
    if (i >= total) return;
    int row = (int)(i >> fshift);
    int f   = (int)(i & ((1u << fshift) - 1u));
    float di = dinv[row];
    out[i] = bias[f] + di * di * h[i];
}

__global__ void gcn_edge_kernel(const float* __restrict__ h,
                                const int* __restrict__ src,
                                const int* __restrict__ dst,
                                const float* __restrict__ dinv,
                                float* __restrict__ out,
                                size_t total, int cshift, int F) {
    size_t i = (size_t)blockIdx.x * blockDim.x + threadIdx.x;
    if (i >= total) return;
    size_t e = i >> cshift;                       // edge id
    int    c = (int)(i & ((1u << cshift) - 1u));  // float4 chunk within row
    int s = src[e], d = dst[e];
    float nrm = dinv[s] * dinv[d];
    const float4 hv = *reinterpret_cast<const float4*>(h + (size_t)s * F + c * 4);
    float* op = out + (size_t)d * F + c * 4;
    atomicAdd(op + 0, nrm * hv.x);
    atomicAdd(op + 1, nrm * hv.y);
    atomicAdd(op + 2, nrm * hv.z);
    atomicAdd(op + 3, nrm * hv.w);
}

// ---------------------------------------------------------------------------
// log_softmax over 64 classes: one wave per row, shuffle reductions (wave32)
// ---------------------------------------------------------------------------
__global__ __launch_bounds__(256) void logsoftmax_kernel(
    const float* __restrict__ in, float* __restrict__ out, int n) {
    int wave = threadIdx.x >> 5, lane = threadIdx.x & 31;
    int row = blockIdx.x * 8 + wave;
    if (row >= n) return;
    const float* p = in + (size_t)row * 64;
    float v0 = p[lane], v1 = p[lane + 32];
    float m = fmaxf(v0, v1);
    for (int o = 16; o; o >>= 1) m = fmaxf(m, __shfl_xor(m, o, 32));
    float s = __expf(v0 - m) + __expf(v1 - m);
    for (int o = 16; o; o >>= 1) s += __shfl_xor(s, o, 32);
    float l = __logf(s);
    out[(size_t)row * 64 + lane]      = v0 - m - l;
    out[(size_t)row * 64 + lane + 32] = v1 - m - l;
}

// ---------------------------------------------------------------------------
// Orchestration
// ---------------------------------------------------------------------------
extern "C" void kernel_launch(void* const* d_in, const int* in_sizes, int n_in,
                              void* d_out, int out_size, void* d_ws, size_t ws_size,
                              hipStream_t stream) {
    const float* x        = (const float*)d_in[0];
    const int*   edge     = (const int*)d_in[1];
    const int*   ego_ids  = (const int*)d_in[2];
    const float* ego_adj  = (const float*)d_in[3];
    const float* normdeg  = (const float*)d_in[4];
    const float* W_ego1   = (const float*)d_in[5];
    const float* b_ego1   = (const float*)d_in[6];
    const float* W_gcn1   = (const float*)d_in[7];
    const float* b_gcn1   = (const float*)d_in[8];
    const float* W_ego2   = (const float*)d_in[9];
    const float* b_ego2   = (const float*)d_in[10];
    const float* W_gcn2   = (const float*)d_in[11];
    const float* b_gcn2   = (const float*)d_in[12];

    const int N = in_sizes[0] / 128;
    const int E = in_sizes[1] / 2;
    const int* src = edge;
    const int* dst = edge + E;

    float* ws   = (float*)d_ws;
    float* deg  = ws;
    float* dinv = ws + N;
    float* B1   = ws + 2 * (size_t)N;              // [N,128]
    float* B2   = B1 + (size_t)N * 128;            // [N,128]
    float* B3   = B2 + (size_t)N * 128;            // [N,128]

    const int T = 256;
    auto cdiv = [](size_t a, size_t b) { return (int)((a + b - 1) / b); };
    const int rowblocks = (N + 15) / 16;
    const int gemm_grid = (rowblocks + 3) / 4;
    const int ego_grid  = (N + 3) / 4;

    // degree normalization
    deg_init_kernel<<<cdiv(N, T), T, 0, stream>>>(deg, N);
    deg_count_kernel<<<cdiv(E, T), T, 0, stream>>>(dst, deg, E);
    deg_rsqrt_kernel<<<cdiv(N, T), T, 0, stream>>>(deg, dinv, N);

    // ---- layer 1 ----------------------------------------------------------
    zero_kernel<<<cdiv((size_t)N * 128, T), T, 0, stream>>>(B1, (size_t)N * 128);
    ego_conv_kernel<<<ego_grid, 128, 0, stream>>>(x, ego_ids, ego_adj, B1, N);
    // h1 = relu((do_conv * norm) @ W_ego1 + b)
    gemm_kernel<128><<<gemm_grid, 128, 0, stream>>>(B1, W_ego1, b_ego1, normdeg, B2, N, 1);
    // hw1 = h1 @ W_gcn1  (bias applied after propagation)
    gemm_kernel<128><<<gemm_grid, 128, 0, stream>>>(B2, W_gcn1, nullptr, nullptr, B1, N, 0);
    // g1 = b_gcn1 + D^-1/2 (A+I) D^-1/2 hw1   -> B2
    gcn_init_kernel<<<cdiv((size_t)N * 128, T), T, 0, stream>>>(B1, b_gcn1, dinv, B2, (size_t)N * 128, 7);
    gcn_edge_kernel<<<cdiv((size_t)E * 32, T), T, 0, stream>>>(B1, src, dst, dinv, B2, (size_t)E * 32, 5, 128);

    // ---- layer 2 ----------------------------------------------------------
    zero_kernel<<<cdiv((size_t)N * 128, T), T, 0, stream>>>(B1, (size_t)N * 128);
    ego_conv_kernel<<<ego_grid, 128, 0, stream>>>(B2, ego_ids, ego_adj, B1, N);
    // h2 = (do_conv * norm) @ W_ego2 + b   (no relu)
    gemm_kernel<128><<<gemm_grid, 128, 0, stream>>>(B1, W_ego2, b_ego2, normdeg, B3, N, 0);
    // hw2 = h2 @ W_gcn2  [N,64]
    gemm_kernel<64><<<gemm_grid, 128, 0, stream>>>(B3, W_gcn2, nullptr, nullptr, B1, N, 0);
    // g2 = b_gcn2 + propagate(hw2)  -> B2 (first N*64 floats)
    gcn_init_kernel<<<cdiv((size_t)N * 64, T), T, 0, stream>>>(B1, b_gcn2, dinv, B2, (size_t)N * 64, 6);
    gcn_edge_kernel<<<cdiv((size_t)E * 16, T), T, 0, stream>>>(B1, src, dst, dinv, B2, (size_t)E * 16, 4, 64);

    // log_softmax -> d_out
    logsoftmax_kernel<<<(N + 7) / 8, 256, 0, stream>>>(B2, (float*)d_out, N);
}
